// RNN_Net3_57775900066583
// MI455X (gfx1250) — compile-verified, matrix operand order checked by reading, not run on hardware
//
#include <hip/hip_runtime.h>

#define N_NODES 100000
#define E_EDGES 3200000

typedef __attribute__((ext_vector_type(16))) _Float16 v16h;
typedef __attribute__((ext_vector_type(8)))  _Float16 v8h;
typedef __attribute__((ext_vector_type(8)))  float    v8f;

// Native f32 atomic add (GLOBAL_ATOMIC_ADD_F32) — avoids CAS-loop lowering.
__device__ __forceinline__ void atomAddF32(float* p, float v) {
    unsafeAtomicAdd(p, v);
}

// ---------------- preprocessing kernels ----------------

// degree counting uses native u32 atomics; buffer reinterpreted as float later
__global__ void k_deg_init(unsigned* __restrict__ deg, int n) {
    int i = blockIdx.x * blockDim.x + threadIdx.x;
    if (i < n) deg[i] = 1u;   // self-loop contributes 1 to every node's degree
}

__global__ void k_deg_count(const int* __restrict__ dst, unsigned* __restrict__ deg, int e) {
    int i = blockIdx.x * blockDim.x + threadIdx.x;
    if (i < e) atomicAdd(&deg[dst[i]], 1u);   // GLOBAL_ATOMIC_ADD_U32
}

// in place: u32 degree -> f32 dinv = rsqrt(deg)
__global__ void k_dinv(unsigned* __restrict__ deg, int n) {
    int i = blockIdx.x * blockDim.x + threadIdx.x;
    if (i < n) {
        float d = (float)deg[i];
        float r = (d > 0.0f) ? rsqrtf(d) : 0.0f;
        ((float*)deg)[i] = r;
    }
}

__global__ void k_norm(const int* __restrict__ src, const int* __restrict__ dst,
                       const float* __restrict__ dinv, float* __restrict__ norm, int e) {
    int i = blockIdx.x * blockDim.x + threadIdx.x;
    if (i < e) norm[i] = dinv[src[i]] * dinv[dst[i]];
}

// Convert W [K x F] (row-major, f32) -> Wt [F x K] (row-major, f16)
__global__ void k_wcvt(const float* __restrict__ w, _Float16* __restrict__ wt, int K, int F) {
    int i = blockIdx.x * blockDim.x + threadIdx.x;
    if (i < K * F) {
        int k = i / F, f = i - k * F;
        wt[f * K + k] = (_Float16)w[i];
    }
}

// fc1: h16[n,f] = f16( x[n,0..2] . fc1_w[:,f] + fc1_b[f] ), f in [0,32)
__global__ void k_fc1(const float* __restrict__ x, const float* __restrict__ w,
                      const float* __restrict__ b, _Float16* __restrict__ h16, int n) {
    int i = blockIdx.x * blockDim.x + threadIdx.x;
    if (i >= n * 32) return;
    int node = i >> 5, f = i & 31;
    float acc = b[f];
#pragma unroll
    for (int k = 0; k < 3; ++k) acc += x[node * 3 + k] * w[k * 32 + f];
    h16[i] = (_Float16)acc;
}

// ---------------- WMMA GEMM: T[N x FOUT] = H16[N x K] * Wt[FOUT x K]^T ----------------
// One wave computes one 16x16 output tile; K in {32,64}; N = 6250 * 16 exactly,
// so EXEC is always all-ones (WMMA requirement). blockDim = (32,2).

template <int K, int FOUT>
__global__ __launch_bounds__(64) void k_gemm_wmma(const _Float16* __restrict__ H,
                                                  const _Float16* __restrict__ Wt,
                                                  float* __restrict__ T) {
    const int lane  = threadIdx.x;            // 0..31
    const int tileM = blockIdx.x * 2 + threadIdx.y;
    const int tileN = blockIdx.y;
    const int r16   = lane & 15;
    const int hi    = lane >> 4;              // 0 or 1
    const int koff  = hi * 8;                 // K-offset for high half-wave (ISA 16-bit A/B layout)

    const _Float16* aBase = H  + (size_t)(tileM * 16 + r16) * K;
    const _Float16* bBase = Wt + (size_t)(tileN * 16 + r16) * K;

    v8f c = {};
#pragma unroll
    for (int kk = 0; kk < K; kk += 32) {
        // lanes 0-15 hold K = {kk+0..7, kk+16..23}; lanes 16-31 hold K = {kk+8..15, kk+24..31}
        v8h a0 = *(const v8h*)(aBase + kk + koff);
        v8h a1 = *(const v8h*)(aBase + kk + 16 + koff);
        v8h b0 = *(const v8h*)(bBase + kk + koff);
        v8h b1 = *(const v8h*)(bBase + kk + 16 + koff);
        v16h a, b;
#pragma unroll
        for (int i = 0; i < 8; ++i) {
            a[i] = a0[i]; a[i + 8] = a1[i];
            b[i] = b0[i]; b[i + 8] = b1[i];
        }
        c = __builtin_amdgcn_wmma_f32_16x16x32_f16(false, a, false, b, (short)0, c, false, false);
    }

    // C/D layout: lane gives column (lane&15); VGPR r gives row r + 8*(lane>=16)
    float* oBase = T + (size_t)(tileM * 16 + hi * 8) * FOUT + tileN * 16 + r16;
#pragma unroll
    for (int r = 0; r < 8; ++r) oBase[(size_t)r * FOUT] = c[r];
}

// ---------------- per-layer glue kernels ----------------

// out = tmp * dinv[node]^2 + bias[f]   (self-loop term + bias; also zero-initializes accumulator)
__global__ void k_init_out(const float* __restrict__ T, const float* __restrict__ dinv,
                           const float* __restrict__ b, float* __restrict__ out,
                           int fout, int total) {
    int i = blockIdx.x * blockDim.x + threadIdx.x;
    if (i >= total) return;
    int node = i / fout, f = i - node * fout;
    float di = dinv[node];
    out[i] = T[i] * di * di + b[f];
}

// Edge scatter: out[dst] += tmp[src] * norm[e], 4 features per thread (float4 gather,
// 4 native f32 atomic adds). Threads [16*k, 16*k+15] share an edge -> coalesced.
template <int FC4>   // FOUT / 4
__global__ void k_scatter(const int* __restrict__ src, const int* __restrict__ dst,
                          const float* __restrict__ norm, const float* __restrict__ T,
                          float* __restrict__ out) {
    unsigned tid = blockIdx.x * blockDim.x + threadIdx.x;
    int e = (int)(tid / FC4);
    int c = (int)(tid % FC4);
    if (e >= E_EDGES) return;
    int   s  = src[e];
    int   d  = dst[e];
    float nr = norm[e];
    float4 v = ((const float4*)T)[(size_t)s * FC4 + c];
    float* o = out + (size_t)d * (FC4 * 4) + c * 4;
    atomAddF32(o + 0, v.x * nr);
    atomAddF32(o + 1, v.y * nr);
    atomAddF32(o + 2, v.z * nr);
    atomAddF32(o + 3, v.w * nr);
}

// ReLU fused with f32 -> f16 conversion for next layer's WMMA input
__global__ void k_relu_cvt(const float* __restrict__ out, _Float16* __restrict__ h16, int total) {
    int i = blockIdx.x * blockDim.x + threadIdx.x;
    if (i < total) {
        float v = out[i];
        h16[i] = (_Float16)(v > 0.0f ? v : 0.0f);
    }
}

// fc2: y[n] = sum_f relu(out[n,f]) * fc2_w[f] + fc2_b   (32 -> 1)
__global__ void k_fc2(const float* __restrict__ h, const float* __restrict__ w,
                      const float* __restrict__ b, float* __restrict__ y, int n) {
    int i = blockIdx.x * blockDim.x + threadIdx.x;
    if (i >= n) return;
    float acc = b[0];
#pragma unroll
    for (int f = 0; f < 32; ++f) {
        float v = h[(size_t)i * 32 + f];
        v = v > 0.0f ? v : 0.0f;
        acc += v * w[f];
    }
    y[i] = acc;
}

// ---------------- host side ----------------

extern "C" void kernel_launch(void* const* d_in, const int* in_sizes, int n_in,
                              void* d_out, int out_size, void* d_ws, size_t ws_size,
                              hipStream_t stream) {
    (void)in_sizes; (void)n_in; (void)out_size; (void)ws_size;

    const float* x     = (const float*)d_in[0];
    const int*   ei    = (const int*)d_in[1];
    const int*   src   = ei;
    const int*   dst   = ei + E_EDGES;
    const float* fc1_w = (const float*)d_in[2];
    const float* fc1_b = (const float*)d_in[3];
    const float* w1    = (const float*)d_in[4];
    const float* b1    = (const float*)d_in[5];
    const float* w2    = (const float*)d_in[6];
    const float* b2    = (const float*)d_in[7];
    const float* w3    = (const float*)d_in[8];
    const float* b3    = (const float*)d_in[9];
    const float* fc2_w = (const float*)d_in[10];
    const float* fc2_b = (const float*)d_in[11];
    float*       yout  = (float*)d_out;

    // workspace carve-up (256B aligned)
    char*  ws  = (char*)d_ws;
    size_t off = 0;
    auto carve = [&](size_t bytes) -> void* {
        void* p = ws + off;
        off += (bytes + 255) & ~(size_t)255;
        return p;
    };
    float*    dinv = (float*)carve((size_t)N_NODES * 4);        // u32 degrees, then f32 dinv
    float*    norm = (float*)carve((size_t)E_EDGES * 4);
    float*    tmp  = (float*)carve((size_t)N_NODES * 64 * 4);
    float*    outb = (float*)carve((size_t)N_NODES * 64 * 4);
    _Float16* h16  = (_Float16*)carve((size_t)N_NODES * 64 * 2);
    _Float16* w1t  = (_Float16*)carve(32 * 64 * 2);
    _Float16* w2t  = (_Float16*)carve(64 * 64 * 2);
    _Float16* w3t  = (_Float16*)carve(64 * 32 * 2);

    const int TPB = 256;
    auto blocks = [](long long total, int tpb) { return (unsigned)((total + tpb - 1) / tpb); };

    // ---- graph preprocessing ----
    k_deg_init <<<blocks(N_NODES, TPB), TPB, 0, stream>>>((unsigned*)dinv, N_NODES);
    k_deg_count<<<blocks(E_EDGES, TPB), TPB, 0, stream>>>(dst, (unsigned*)dinv, E_EDGES);
    k_dinv     <<<blocks(N_NODES, TPB), TPB, 0, stream>>>((unsigned*)dinv, N_NODES);
    k_norm     <<<blocks(E_EDGES, TPB), TPB, 0, stream>>>(src, dst, dinv, norm, E_EDGES);

    // ---- weight conversion (f32 -> f16, transposed) ----
    k_wcvt<<<blocks(32 * 64, TPB), TPB, 0, stream>>>(w1, w1t, 32, 64);
    k_wcvt<<<blocks(64 * 64, TPB), TPB, 0, stream>>>(w2, w2t, 64, 64);
    k_wcvt<<<blocks(64 * 32, TPB), TPB, 0, stream>>>(w3, w3t, 64, 32);

    // ---- fc1 ----
    k_fc1<<<blocks((long long)N_NODES * 32, TPB), TPB, 0, stream>>>(x, fc1_w, fc1_b, h16, N_NODES);

    const dim3 gemmBlk(32, 2);   // 2 waves per block, one 16x16 tile each

    for (int d = 0; d < 4; ++d) {
        // gcn layer 1: 32 -> 64
        k_gemm_wmma<32, 64><<<dim3(3125, 4), gemmBlk, 0, stream>>>(h16, w1t, tmp);
        k_init_out <<<blocks((long long)N_NODES * 64, TPB), TPB, 0, stream>>>(tmp, dinv, b1, outb, 64, N_NODES * 64);
        k_scatter<16><<<blocks((long long)E_EDGES * 16, TPB), TPB, 0, stream>>>(src, dst, norm, tmp, outb);
        k_relu_cvt <<<blocks((long long)N_NODES * 64, TPB), TPB, 0, stream>>>(outb, h16, N_NODES * 64);

        // gcn layer 2: 64 -> 64
        k_gemm_wmma<64, 64><<<dim3(3125, 4), gemmBlk, 0, stream>>>(h16, w2t, tmp);
        k_init_out <<<blocks((long long)N_NODES * 64, TPB), TPB, 0, stream>>>(tmp, dinv, b2, outb, 64, N_NODES * 64);
        k_scatter<16><<<blocks((long long)E_EDGES * 16, TPB), TPB, 0, stream>>>(src, dst, norm, tmp, outb);
        k_relu_cvt <<<blocks((long long)N_NODES * 64, TPB), TPB, 0, stream>>>(outb, h16, N_NODES * 64);

        // gcn layer 3: 64 -> 32
        k_gemm_wmma<64, 32><<<dim3(3125, 2), gemmBlk, 0, stream>>>(h16, w3t, tmp);
        k_init_out <<<blocks((long long)N_NODES * 32, TPB), TPB, 0, stream>>>(tmp, dinv, b3, outb, 32, N_NODES * 32);
        k_scatter<8><<<blocks((long long)E_EDGES * 8, TPB), TPB, 0, stream>>>(src, dst, norm, tmp, outb);
        if (d < 3) {  // final layer's h16 is unused; fc2 reads the f32 accumulator
            k_relu_cvt<<<blocks((long long)N_NODES * 32, TPB), TPB, 0, stream>>>(outb, h16, N_NODES * 32);
        }
    }

    // ---- fc2 (relu applied inline from f32 accumulator) ----
    k_fc2<<<blocks(N_NODES, TPB), TPB, 0, stream>>>(outb, fc2_w, fc2_b, yout, N_NODES);
}
